// TinyLSTM_72988674228333
// MI455X (gfx1250) — matrix operand order repaired and checked by reading, not compiled, not run on hardware
//
#include <hip/hip_runtime.h>
#include <hip/hip_bf16.h>

// ---------------------------------------------------------------------------
// TinyLSTM for MI455X (gfx1250): bf16 WMMA recurrence + bf16 WMMA logits GEMM
// ---------------------------------------------------------------------------

typedef __attribute__((ext_vector_type(16))) __bf16        v16bf;
typedef __attribute__((ext_vector_type(8)))  float         v8f;
typedef __attribute__((ext_vector_type(4)))  unsigned int  u32x4;
typedef __attribute__((ext_vector_type(4)))  float         f32x4;

#define BB 64
#define TT 2048
#define HH 256
#define GG 1024   /* 4*H */
#define VV 256

#define MR 32     /* batch rows per workgroup (2 M-tiles) */
#define GP 1028   /* gates LDS pitch (floats), padded vs 1024 to break bank conflicts */
#define HP 264    /* h LDS pitch (bf16 elems), padded vs 256 */

union FragBF { u32x4 q[2]; v16bf v; };
union Half8  { u32x4 q;    __bf16 h[8]; };

__device__ __forceinline__ float fsigmoid(float x) { return 1.0f / (1.0f + __expf(-x)); }
__device__ __forceinline__ float ftanh(float x)    { return 2.0f / (1.0f + __expf(-2.0f * x)) - 1.0f; }

// ---- prep: Wib[token][g] = W_ih[g][token] + b_ih[g] + b_hh[g] (f32, 1MB, L2-resident)
__global__ void prep_wib(const float* __restrict__ W_ih, const float* __restrict__ b_ih,
                         const float* __restrict__ b_hh, float* __restrict__ wib) {
  int g = blockIdx.x;    // 0..1023
  int v = threadIdx.x;   // 0..255
  wib[v * GG + g] = W_ih[g * VV + v] + b_ih[g] + b_hh[g];
}

// ---- prep: f32 -> bf16 conversion (W_hh, fc_W)
__global__ void prep_bf16(const float* __restrict__ src, __bf16* __restrict__ dst, int n) {
  int i = blockIdx.x * blockDim.x + threadIdx.x;
  if (i < n) dst[i] = (__bf16)src[i];
}

// ---------------------------------------------------------------------------
// Recurrence: 2 blocks x 512 threads (16 waves, 4/SIMD). Block handles 32
// batch rows (2 M-tiles) so every streamed W_hh B-fragment feeds TWO WMMAs:
// halves L2 traffic per step and doubles math per load.
// Phase 1: gates[32,1024] = h_bf16[32,256] @ Whh_bf16^T; wave w owns 64 gate
//          columns (4 n-tiles) x 2 m-tiles; K-loop of 8 x 16x16x32 bf16.
//          Laundered zero *offset* (not pointer: that would lose the global
//          address space and emit flat_load) keeps B loads in-loop, no spill.
// Phase 2: thread (row m=tid&31, 16 contiguous j's) applies i/f/g/o
//          activations, updates c (registers), writes h to LDS + hs global.
// ---------------------------------------------------------------------------
__global__ void __launch_bounds__(512)
lstm_kernel(const int* __restrict__ x, const float* __restrict__ wib,
            const __bf16* __restrict__ whh, __bf16* __restrict__ hs,
            float* __restrict__ hn, float* __restrict__ cn) {
  extern __shared__ char smem[];
  float*  gates = (float*)smem;                    // [MR][GP] f32
  __bf16* hbf   = (__bf16*)(smem + MR * GP * 4);   // [MR][HP] bf16
  char*   hbase = smem + MR * GP * 4;

  const int tid  = threadIdx.x;
  const int lane = tid & 31;
  const int wave = tid >> 5;    // 0..15
  const int m_a  = lane & 15;   // A-matrix row / B,C,D column lane
  const int kh   = lane >> 4;   // lane half selects K sub-range / C row offset

  // zero initial hidden state in LDS
  for (int i = tid; i < MR * HP; i += 512) hbf[i] = (__bf16)0.0f;

  // phase-2 assignment: row m2 (0..31), 16 contiguous h-columns from jb
  const int m2 = tid & 31;
  const int jb = (tid >> 5) * 16;    // 0..240
  const int b  = blockIdx.x * MR + m2;

  float cst[16], hlast[16];
#pragma unroll
  for (int k = 0; k < 16; k++) { cst[k] = 0.0f; hlast[k] = 0.0f; }

  const int nbase = wave * 64;                        // 4 n-tiles per wave
  const int aoff0 = m_a * (HP * 2) + kh * 16;         // A m-tile 0
  const int aoff1 = (16 + m_a) * (HP * 2) + kh * 16;  // A m-tile 1

  __syncthreads();

  for (int t = 0; t < TT; t++) {
    // Opaque zero offset: keeps Whh loads inside the loop (no LICM -> no
    // spill) while preserving the global address space (global_load_b128).
    unsigned int zofs = 0;
    asm volatile("" : "+s"(zofs));
    const __bf16* whh_t = whh + zofs;

    // -------- phase 1: WMMA gates (2 m-tiles x 4 n-tiles) --------
    v8f acc0[4], acc1[4];
#pragma unroll
    for (int nt = 0; nt < 4; nt++)
#pragma unroll
      for (int r = 0; r < 8; r++) { acc0[nt][r] = 0.0f; acc1[nt][r] = 0.0f; }

#pragma unroll
    for (int kt = 0; kt < 8; kt++) {
      // A fragments: two 16x32 bf16 h tiles from LDS (2x two ds_load_b128)
      FragBF a0, a1;
      const char* ap0 = hbase + aoff0 + kt * 64;
      const char* ap1 = hbase + aoff1 + kt * 64;
      a0.q[0] = *(const u32x4*)(ap0);
      a0.q[1] = *(const u32x4*)(ap0 + 32);
      a1.q[0] = *(const u32x4*)(ap1);
      a1.q[1] = *(const u32x4*)(ap1 + 32);
#pragma unroll
      for (int nt = 0; nt < 4; nt++) {
        // B fragment: column n of Whh^T == row n of Whh; two 16-byte chunks (k 8h..8h+7, 16+8h..)
        const int ncol = nbase + nt * 16 + m_a;
        const char* bp = (const char*)whh_t + ncol * (HH * 2) + kt * 64 + kh * 16;
        FragBF bfrag;
        bfrag.q[0] = *(const u32x4*)(bp);
        bfrag.q[1] = *(const u32x4*)(bp + 32);
        acc0[nt] = __builtin_amdgcn_wmma_f32_16x16x32_bf16(
            false, a0.v, false, bfrag.v, (short)0, acc0[nt], false, false);
        acc1[nt] = __builtin_amdgcn_wmma_f32_16x16x32_bf16(
            false, a1.v, false, bfrag.v, (short)0, acc1[nt], false, false);
      }
    }
    // scatter accumulators: VGPR r -> row (mt*16 + r + kh*8), col ncol
#pragma unroll
    for (int nt = 0; nt < 4; nt++) {
      const int ncol = nbase + nt * 16 + m_a;
#pragma unroll
      for (int r = 0; r < 8; r++) {
        gates[(r + kh * 8) * GP + ncol]      = acc0[nt][r];
        gates[(16 + r + kh * 8) * GP + ncol] = acc1[nt][r];
      }
    }
    __syncthreads();

    // -------- phase 2: activations + state update --------
    const int tok = x[(size_t)b * TT + t];
    const float* wibrow = wib + (size_t)tok * GG;

    // Wib (global/L2, long latency): hoist all 16 f32x4 loads up front.
    f32x4 wiv[4], wfv[4], wgv[4], wov[4];
#pragma unroll
    for (int q = 0; q < 4; q++) {
      wiv[q] = *(const f32x4*)(wibrow + jb + 4 * q);
      wfv[q] = *(const f32x4*)(wibrow + 256 + jb + 4 * q);
      wgv[q] = *(const f32x4*)(wibrow + 512 + jb + 4 * q);
      wov[q] = *(const f32x4*)(wibrow + 768 + jb + 4 * q);
    }

    Half8 hout[2];
#pragma unroll
    for (int q = 0; q < 4; q++) {
      // gates (LDS, short latency): consume per 4-element chunk
      const f32x4 gi = *(const f32x4*)(gates + m2 * GP + jb + 4 * q);
      const f32x4 gf = *(const f32x4*)(gates + m2 * GP + 256 + jb + 4 * q);
      const f32x4 gg = *(const f32x4*)(gates + m2 * GP + 512 + jb + 4 * q);
      const f32x4 go = *(const f32x4*)(gates + m2 * GP + 768 + jb + 4 * q);
#pragma unroll
      for (int e = 0; e < 4; e++) {
        const int k = q * 4 + e;
        float iv = fsigmoid(gi[e] + wiv[q][e]);
        float fv = fsigmoid(gf[e] + wfv[q][e]);
        float gv = ftanh(gg[e] + wgv[q][e]);
        float ov = fsigmoid(go[e] + wov[q][e]);
        const float cv = fv * cst[k] + iv * gv;
        cst[k] = cv;
        const float hv = ov * ftanh(cv);
        hlast[k] = hv;
        hout[k >> 3].h[k & 7] = (__bf16)hv;
      }
    }
    // new h -> LDS (two ds_store_b128)
    {
      char* hp = hbase + m2 * (HP * 2) + jb * 2;
      *(u32x4*)(hp)      = hout[0].q;
      *(u32x4*)(hp + 16) = hout[1].q;
    }
    // hidden state -> global hs (bf16), for the logits GEMM
    {
      u32x4* dst = (u32x4*)(hs + ((size_t)b * TT + t) * HH + jb);
      dst[0] = hout[0].q;
      dst[1] = hout[1].q;
    }
    __syncthreads();
  }

  // final h_n, c_n (f32)
#pragma unroll
  for (int k = 0; k < 16; k++) {
    const int j = jb + k;
    hn[b * HH + j] = hlast[k];
    cn[b * HH + j] = cst[k];
  }
}

// ---------------------------------------------------------------------------
// Logits: [B*T,256] = hs_bf16 @ fcW_bf16^T + fc_b.  One block = 16 rows x 256
// cols; 8 waves x 2 n-tiles each; K = 256 in 8 x 16x16x32 bf16 WMMA steps.
// Memory-bound: ~192 MB total traffic ≈ 8 us at 23.3 TB/s.
// ---------------------------------------------------------------------------
__global__ void __launch_bounds__(256)
logits_kernel(const __bf16* __restrict__ hs, const __bf16* __restrict__ fcw,
              const float* __restrict__ fcb, float* __restrict__ out) {
  const int tid  = threadIdx.x;
  const int lane = tid & 31;
  const int wave = tid >> 5;
  const int m_a  = lane & 15;
  const int kh   = lane >> 4;
  const size_t mbase = (size_t)blockIdx.x * 16;
  const int nb0 = wave * 32;

  v8f acc[2];
  const float f0 = fcb[nb0 + m_a];
  const float f1 = fcb[nb0 + 16 + m_a];
#pragma unroll
  for (int r = 0; r < 8; r++) { acc[0][r] = f0; acc[1][r] = f1; }

#pragma unroll
  for (int kt = 0; kt < 8; kt++) {
    FragBF a;
    const char* ap = (const char*)(hs + (mbase + m_a) * HH) + kt * 64 + kh * 16;
    a.q[0] = *(const u32x4*)(ap);
    a.q[1] = *(const u32x4*)(ap + 32);
#pragma unroll
    for (int nt = 0; nt < 2; nt++) {
      const int ncol = nb0 + nt * 16 + m_a;
      const char* bp = (const char*)(fcw + ncol * HH) + kt * 64 + kh * 16;
      FragBF bfrag;
      bfrag.q[0] = *(const u32x4*)(bp);
      bfrag.q[1] = *(const u32x4*)(bp + 32);
      acc[nt] = __builtin_amdgcn_wmma_f32_16x16x32_bf16(
          false, a.v, false, bfrag.v, (short)0, acc[nt], false, false);
    }
  }
#pragma unroll
  for (int nt = 0; nt < 2; nt++) {
    const int ncol = nb0 + nt * 16 + m_a;
#pragma unroll
    for (int r = 0; r < 8; r++) {
      const size_t row = mbase + r + kh * 8;
      out[row * 256 + ncol] = acc[nt][r];
    }
  }
}

// ---------------------------------------------------------------------------
extern "C" void kernel_launch(void* const* d_in, const int* in_sizes, int n_in,
                              void* d_out, int out_size, void* d_ws, size_t ws_size,
                              hipStream_t stream) {
  const int*   x    = (const int*)d_in[0];
  const float* W_ih = (const float*)d_in[1];
  const float* W_hh = (const float*)d_in[2];
  const float* b_ih = (const float*)d_in[3];
  const float* b_hh = (const float*)d_in[4];
  const float* fc_W = (const float*)d_in[5];
  const float* fc_b = (const float*)d_in[6];

  // workspace layout
  char* ws = (char*)d_ws;
  float*  wib = (float*)(ws);                                      // 1 MB
  __bf16* whh = (__bf16*)(ws + (1u << 20));                        // 512 KB
  __bf16* fcw = (__bf16*)(ws + (1u << 20) + (512u << 10));         // 128 KB
  __bf16* hs  = (__bf16*)(ws + (1u << 20) + (640u << 10));         // 64 MB

  float* out = (float*)d_out;
  float* hn  = out + (size_t)BB * TT * VV;   // logits are [B,T,256]
  float* cn  = hn + (size_t)BB * HH;

  prep_wib<<<GG, VV, 0, stream>>>(W_ih, b_ih, b_hh, wib);
  prep_bf16<<<(GG * HH + 255) / 256, 256, 0, stream>>>(W_hh, whh, GG * HH);
  prep_bf16<<<(VV * HH + 255) / 256, 256, 0, stream>>>(fc_W, fcw, VV * HH);

  const size_t lds_bytes = (size_t)MR * GP * 4 + (size_t)MR * HP * 2;  // ~145 KB
  lstm_kernel<<<BB / MR, 512, lds_bytes, stream>>>(x, wib, whh, hs, hn, cn);

  logits_kernel<<<(BB * TT) / 16, 256, 0, stream>>>(hs, fcw, fc_b, out);
}
